// FastGTNs_40948218200524
// MI455X (gfx1250) — compile-verified, matrix operand order checked by reading, not run on hardware
//
#include <hip/hip_runtime.h>
#include <hip/hip_bf16.h>

typedef __attribute__((ext_vector_type(16))) __bf16 v16bf;
typedef __attribute__((ext_vector_type(8)))  float  v8f;

constexpr int   NNODES = 100000;
constexpr int   KIN    = 256;
constexpr int   DOUT   = 128;
constexpr int   CCH    = 2;
constexpr int   ETY    = 5;
constexpr int   NEDGE  = 500000;
constexpr float BETA_C = 0.1f;

union Frag16 {
    v16bf bf;
    unsigned short u[16];
    uint4 q[2];
};

__device__ __forceinline__ unsigned short f2bf(float f) {
    unsigned int x = __float_as_uint(f);
    x += 0x7FFFu + ((x >> 16) & 1u);   // round-to-nearest-even
    return (unsigned short)(x >> 16);
}

// ---- transpose+convert Ws[c] (KIN x DOUT, f32) -> WsT[c] (DOUT x KIN, bf16) ----
__global__ void prep_wT(const float* __restrict__ Ws, unsigned short* __restrict__ WsT) {
    const int total = CCH * DOUT * KIN;
    for (int i = blockIdx.x * blockDim.x + threadIdx.x; i < total;
         i += gridDim.x * blockDim.x) {
        int c   = i / (DOUT * KIN);
        int rem = i % (DOUT * KIN);
        int n   = rem / KIN;
        int k   = rem % KIN;
        WsT[i] = f2bf(Ws[((size_t)c * KIN + k) * DOUT + n]);
    }
}

// ---- softmax filters, folded W2 = lin1_w @ lin2_w (256x3), folded bias ----
__global__ void prep_small(const float* __restrict__ conv_w,
                           const float* __restrict__ lin1_w,
                           const float* __restrict__ lin1_b,
                           const float* __restrict__ lin2_w,
                           const float* __restrict__ lin2_b,
                           float* __restrict__ filt,
                           float* __restrict__ W2,
                           float* __restrict__ b2) {
    int t = threadIdx.x;
    if (t < 2 * CCH) {                         // (l,c) pairs: L*C = 4 softmaxes over ETY
        const float* cw = conv_w + t * ETY;
        float mx = cw[0];
        for (int j = 1; j < ETY; j++) mx = fmaxf(mx, cw[j]);
        float ex[ETY]; float s = 0.f;
        for (int j = 0; j < ETY; j++) { ex[j] = __expf(cw[j] - mx); s += ex[j]; }
        for (int j = 0; j < ETY; j++) filt[t * ETY + j] = ex[j] / s;
    }
    if (t < 2 * DOUT) {                        // one row of W2 per thread (256 rows)
        float s0 = 0.f, s1 = 0.f, s2 = 0.f;
        const float* lr = lin1_w + (size_t)t * DOUT;
        for (int m = 0; m < DOUT; m++) {
            float v = lr[m];
            const float* l2 = lin2_w + m * 3;
            s0 += v * l2[0]; s1 += v * l2[1]; s2 += v * l2[2];
        }
        W2[t * 3 + 0] = s0; W2[t * 3 + 1] = s1; W2[t * 3 + 2] = s2;
    }
    if (t < 3) {                               // b2 = lin1_b @ lin2_w + lin2_b
        float s = lin2_b[t];
        for (int m = 0; m < DOUT; m++) s += lin1_b[m] * lin2_w[m * 3 + t];
        b2[t] = s;
    }
}

// ---- X_ [c] = X @ Ws[c] via v_wmma_f32_16x16x32_bf16; one wave = 16x128 tile ----
__global__ void gemm_proj(const float* __restrict__ X,
                          const unsigned short* __restrict__ WsT,
                          float* __restrict__ Xp) {
    const int wave  = threadIdx.x >> 5;
    const int lane  = threadIdx.x & 31;
    const int mtile = blockIdx.x * 8 + wave;
    if (mtile >= NNODES / 16) return;
    const int c  = blockIdx.y;
    const int lt = lane & 15;
    const int hi = lane >> 4;
    const size_t M = (size_t)mtile * 16 + lt;
    const unsigned short* Wc = WsT + (size_t)c * DOUT * KIN;

    v8f acc[8];
    v8f z = {0.f,0.f,0.f,0.f,0.f,0.f,0.f,0.f};
    #pragma unroll
    for (int i = 0; i < 8; i++) acc[i] = z;

    for (int k0 = 0; k0 < KIN; k0 += 32) {
        const int kb = k0 + hi * 8;
        const float* xp = X + M * KIN + kb;
        Frag16 a;
        float4 x0 = *(const float4*)(xp);
        float4 x1 = *(const float4*)(xp + 4);
        float4 x2 = *(const float4*)(xp + 16);
        float4 x3 = *(const float4*)(xp + 20);
        a.u[0]=f2bf(x0.x);  a.u[1]=f2bf(x0.y);  a.u[2]=f2bf(x0.z);  a.u[3]=f2bf(x0.w);
        a.u[4]=f2bf(x1.x);  a.u[5]=f2bf(x1.y);  a.u[6]=f2bf(x1.z);  a.u[7]=f2bf(x1.w);
        a.u[8]=f2bf(x2.x);  a.u[9]=f2bf(x2.y);  a.u[10]=f2bf(x2.z); a.u[11]=f2bf(x2.w);
        a.u[12]=f2bf(x3.x); a.u[13]=f2bf(x3.y); a.u[14]=f2bf(x3.z); a.u[15]=f2bf(x3.w);
        #pragma unroll
        for (int nt = 0; nt < 8; nt++) {
            Frag16 b;
            const unsigned short* bp = Wc + (size_t)(nt * 16 + lt) * KIN + kb;
            b.q[0] = *(const uint4*)(bp);
            b.q[1] = *(const uint4*)(bp + 16);
            acc[nt] = __builtin_amdgcn_wmma_f32_16x16x32_bf16(
                false, a.bf, false, b.bf, (short)0, acc[nt], false, false);
        }
    }
    float* outb = Xp + (size_t)c * NNODES * DOUT;
    #pragma unroll
    for (int nt = 0; nt < 8; nt++) {
        #pragma unroll
        for (int r = 0; r < 8; r++) {
            outb[((size_t)mtile * 16 + r + 8 * hi) * DOUT + nt * 16 + lt] = acc[nt][r];
        }
    }
}

__global__ void zero_f4(float4* __restrict__ p, long n4) {
    long stride = (long)gridDim.x * blockDim.x;
    float4 zz = {0.f, 0.f, 0.f, 0.f};
    for (long i = (long)blockIdx.x * blockDim.x + threadIdx.x; i < n4; i += stride)
        p[i] = zz;
}

// ---- one wave per edge (x4 edges/wave); lane = float4 slice of 128-wide row ----
__global__ void spmm_layer(const int* __restrict__ eidx,
                           const float* __restrict__ eval,
                           const float* __restrict__ filt,   // this layer: [C][ETY]
                           const float* __restrict__ H,
                           float* __restrict__ Hn) {
    const int j    = blockIdx.y;
    const int wave = threadIdx.x >> 5;
    const int lane = threadIdx.x & 31;
    const int*   rowp = eidx + (size_t)j * 2 * NEDGE;
    const int*   colp = rowp + NEDGE;
    const float* valp = eval + (size_t)j * NEDGE;
    const float f0 = filt[0 * ETY + j];
    const float f1 = filt[1 * ETY + j];
    const int ebase = (blockIdx.x * 8 + wave) * 4;
    #pragma unroll
    for (int it = 0; it < 4; it++) {
        const int e   = ebase + it;
        const int row = rowp[e];
        const int col = colp[e];
        const float v = valp[e];
        const float w0 = v * f0;
        const float w1 = v * f1;
        {   // channel 0
            float4 m = ((const float4*)(H + (size_t)col * DOUT))[lane];
            float* d = Hn + (size_t)row * DOUT + lane * 4;
            unsafeAtomicAdd(d + 0, m.x * w0);
            unsafeAtomicAdd(d + 1, m.y * w0);
            unsafeAtomicAdd(d + 2, m.z * w0);
            unsafeAtomicAdd(d + 3, m.w * w0);
        }
        {   // channel 1
            float4 m = ((const float4*)(H + ((size_t)NNODES + col) * DOUT))[lane];
            float* d = Hn + ((size_t)NNODES + row) * DOUT + lane * 4;
            unsafeAtomicAdd(d + 0, m.x * w1);
            unsafeAtomicAdd(d + 1, m.y * w1);
            unsafeAtomicAdd(d + 2, m.z * w1);
            unsafeAtomicAdd(d + 3, m.w * w1);
        }
    }
}

// ---- fused relu(beta*X_ + (1-beta)*H), concat channels, @W2(256x3) + b2 ----
__global__ void final_out(const float* __restrict__ Xp,
                          const float* __restrict__ H,
                          const float* __restrict__ W2,
                          const float* __restrict__ b2,
                          float* __restrict__ out) {
    const int wave = threadIdx.x >> 5;
    const int lane = threadIdx.x & 31;
    const long n = (long)blockIdx.x * 8 + wave;
    if (n >= NNODES) return;
    float s0 = 0.f, s1 = 0.f, s2 = 0.f;
    #pragma unroll
    for (int i = 0; i < 8; i++) {
        const int d  = i * 32 + lane;     // 0..255 across concat of 2 channels
        const int c  = d >> 7;
        const int dd = d & 127;
        const size_t idx = ((size_t)c * NNODES + n) * DOUT + dd;
        float h = BETA_C * Xp[idx] + (1.f - BETA_C) * H[idx];
        h = fmaxf(h, 0.f);
        const float* w = W2 + d * 3;
        s0 += h * w[0]; s1 += h * w[1]; s2 += h * w[2];
    }
    #pragma unroll
    for (int off = 16; off > 0; off >>= 1) {
        s0 += __shfl_xor(s0, off, 32);
        s1 += __shfl_xor(s1, off, 32);
        s2 += __shfl_xor(s2, off, 32);
    }
    if (lane == 0) {
        out[n * 3 + 0] = s0 + b2[0];
        out[n * 3 + 1] = s1 + b2[1];
        out[n * 3 + 2] = s2 + b2[2];
    }
}

extern "C" void kernel_launch(void* const* d_in, const int* in_sizes, int n_in,
                              void* d_out, int out_size, void* d_ws, size_t ws_size,
                              hipStream_t stream) {
    (void)in_sizes; (void)n_in; (void)out_size; (void)ws_size;
    const int*   edge_index = (const int*)d_in[0];
    const float* edge_value = (const float*)d_in[1];
    const float* X          = (const float*)d_in[2];
    const float* Ws         = (const float*)d_in[3];
    const float* conv_w     = (const float*)d_in[4];
    const float* lin1_w     = (const float*)d_in[5];
    const float* lin1_b     = (const float*)d_in[6];
    const float* lin2_w     = (const float*)d_in[7];
    const float* lin2_b     = (const float*)d_in[8];
    float* out = (float*)d_out;

    // workspace layout (256B-aligned regions)
    char* ws = (char*)d_ws;
    unsigned short* WsT = (unsigned short*)(ws + 0);                  // 131072 B
    float* filt = (float*)(ws + 131072);                              // 80 B (pad 256)
    float* W2   = (float*)(ws + 131328);                              // 3072 B
    float* b2   = (float*)(ws + 134400);                              // 12 B (pad 256)
    float* Xp   = (float*)(ws + 134656);                              // 102,400,000 B
    float* Ha   = (float*)(ws + 134656 + 102400000L);
    float* Hb   = (float*)(ws + 134656 + 204800000L);

    prep_wT<<<64, 256, 0, stream>>>(Ws, WsT);
    prep_small<<<1, 256, 0, stream>>>(conv_w, lin1_w, lin1_b, lin2_w, lin2_b,
                                      filt, W2, b2);

    gemm_proj<<<dim3((NNODES / 16 + 7) / 8, CCH), 256, 0, stream>>>(X, WsT, Xp);

    const long n4 = (long)CCH * NNODES * DOUT / 4;
    zero_f4<<<4096, 256, 0, stream>>>((float4*)Ha, n4);
    spmm_layer<<<dim3(NEDGE / 32, ETY), 256, 0, stream>>>(
        edge_index, edge_value, filt + 0, Xp, Ha);
    zero_f4<<<4096, 256, 0, stream>>>((float4*)Hb, n4);
    spmm_layer<<<dim3(NEDGE / 32, ETY), 256, 0, stream>>>(
        edge_index, edge_value, filt + CCH * ETY, Ha, Hb);

    final_out<<<(NNODES + 7) / 8, 256, 0, stream>>>(Xp, Hb, W2, b2, out);
}
